// XorzoTransformer_17136919511711
// MI455X (gfx1250) — matrix operand order repaired and checked by reading, not compiled, not run on hardware
//
#include <hip/hip_runtime.h>
#include <hip/hip_bf16.h>
#include <math.h>

typedef _Float16 half_t;
typedef __attribute__((ext_vector_type(16))) _Float16 v16h;
typedef __attribute__((ext_vector_type(8)))  _Float16 v8h;
typedef __attribute__((ext_vector_type(8)))  float    v8f;

#define BB 4
#define TT 1024
#define DD 1024
#define HH 16
#define DH 64
#define MM (BB*TT)        // 4096 rows
#define QT 16             // query tile

__device__ __forceinline__ float sigmoidf_(float v) { return 1.0f / (1.0f + __expf(-v)); }

// ---------------------------------------------------------------- f32 -> f16
__global__ void f2h_kernel(const float* __restrict__ src, half_t* __restrict__ dst, int n) {
    int i = blockIdx.x * blockDim.x + threadIdx.x;
    if (i < n) dst[i] = (half_t)src[i];
}

// --------------------------------------- f32 RxC -> f16 CxR (transposed copy)
__global__ void f2h_T_kernel(const float* __restrict__ src, half_t* __restrict__ dst,
                             int R, int C) {
    int idx = blockIdx.x * blockDim.x + threadIdx.x;
    if (idx >= R * C) return;
    int r = idx / C, c = idx % C;
    dst[(size_t)c * R + r] = (half_t)src[idx];
}

// ------------------------------------------------- aperture / expression gates
__global__ void gates_kernel(const float* __restrict__ x,
                             const float* __restrict__ Wa, const float* __restrict__ ba,
                             const float* __restrict__ We, const float* __restrict__ be,
                             float* __restrict__ ap, float* __restrict__ ex) {
    int idx = blockIdx.x * blockDim.x + threadIdx.x;          // over MM*HH
    if (idx >= MM * HH) return;
    int row = idx / HH, h = idx % HH;
    const float* xr = x + (size_t)row * DD;
    float sa = ba[h], se = be[h];
    for (int k = 0; k < DD; ++k) {
        float xv = xr[k];
        sa += xv * Wa[(size_t)k * HH + h];
        se += xv * We[(size_t)k * HH + h];
    }
    ap[idx] = sigmoidf_(sa);
    ex[idx] = sigmoidf_(se);
}

// ------------------------------------------------------ WMMA fragment loaders
// A: f16 row-major [16 x Kslice]; contiguous K per lane half (k = 16h + s), matching B.
__device__ __forceinline__ v16h load_a_frag(const half_t* __restrict__ Arow, size_t stride,
                                            int k0, int lane) {
    int l15 = lane & 15;
    int ko  = (lane < 16) ? 0 : 16;
    const v8h* a0 = (const v8h*)(Arow + (size_t)l15 * stride + k0 + ko);
    const v8h* a1 = (const v8h*)(Arow + (size_t)l15 * stride + k0 + 8 + ko);
    v8h alo = a0[0], ahi = a1[0];
    v16h a;
#pragma unroll
    for (int i = 0; i < 8; ++i) { a[i] = alo[i]; a[8 + i] = ahi[i]; }
    return a;
}
// B from K-contiguous (transposed) storage: lane covers col n, halves are K-contiguous.
__device__ __forceinline__ v16h load_b_fragT(const half_t* __restrict__ Wt, size_t stride,
                                             int n, int k0, int lane) {
    int kb = k0 + ((lane < 16) ? 0 : 16);
    return *(const v16h*)(Wt + (size_t)n * stride + kb);
}

// ------------------- X(f16)[MMxDD] @ Wt(f16, transposed NxK) -> 16x64 per wave
// mode 0: write (B,H,T,dh) f16    mode 1: write (B,H,dh,T) f16 (for V)
__global__ void gemm16x64_heads_kernel(const half_t* __restrict__ A,
                                       const half_t* __restrict__ Wt,
                                       half_t* __restrict__ outp, int mode) {
    int wave = threadIdx.x >> 5, lane = threadIdx.x & 31;
    const int tilesN = DD >> 6;                               // 64-wide strips
    int tile = blockIdx.x * 8 + wave;
    if (tile >= (MM >> 4) * tilesN) return;                   // wave-uniform
    int tm = tile / tilesN, tn = tile % tilesN;
    int l15 = lane & 15;
    const half_t* Arow = A + (size_t)tm * 16 * DD;
    v8f c0 = {}, c1 = {}, c2 = {}, c3 = {};
    for (int k0 = 0; k0 < DD; k0 += 32) {
        v16h a = load_a_frag(Arow, DD, k0, lane);
        v16h b0 = load_b_fragT(Wt, DD, tn * 64 + 0 * 16 + l15, k0, lane);
        v16h b1 = load_b_fragT(Wt, DD, tn * 64 + 1 * 16 + l15, k0, lane);
        v16h b2 = load_b_fragT(Wt, DD, tn * 64 + 2 * 16 + l15, k0, lane);
        v16h b3 = load_b_fragT(Wt, DD, tn * 64 + 3 * 16 + l15, k0, lane);
        c0 = __builtin_amdgcn_wmma_f32_16x16x32_f16(false, a, false, b0, (short)0, c0, false, false);
        c1 = __builtin_amdgcn_wmma_f32_16x16x32_f16(false, a, false, b1, (short)0, c1, false, false);
        c2 = __builtin_amdgcn_wmma_f32_16x16x32_f16(false, a, false, b2, (short)0, c2, false, false);
        c3 = __builtin_amdgcn_wmma_f32_16x16x32_f16(false, a, false, b3, (short)0, c3, false, false);
    }
    v8f cs[4] = {c0, c1, c2, c3};
#pragma unroll
    for (int j = 0; j < 4; ++j) {
        int col = tn * 64 + j * 16 + l15;
        int h = col / DH, d = col % DH;
#pragma unroll
        for (int r = 0; r < 8; ++r) {
            int row = tm * 16 + r + ((lane < 16) ? 0 : 8);
            int b = row / TT, t = row % TT;
            size_t o = (mode == 0)
                     ? ((((size_t)b * HH + h) * TT + t) * DH + d)
                     : ((((size_t)b * HH + h) * DH + d) * TT + t);
            outp[o] = (half_t)cs[j][r];
        }
    }
}

// ------------------- combined(f16) @ WoutT(f16) -> f32 row-major output
__global__ void gemm16x64_out_kernel(const half_t* __restrict__ A,
                                     const half_t* __restrict__ Wt,
                                     float* __restrict__ outF) {
    int wave = threadIdx.x >> 5, lane = threadIdx.x & 31;
    const int tilesN = DD >> 6;
    int tile = blockIdx.x * 8 + wave;
    if (tile >= (MM >> 4) * tilesN) return;
    int tm = tile / tilesN, tn = tile % tilesN;
    int l15 = lane & 15;
    const half_t* Arow = A + (size_t)tm * 16 * DD;
    v8f c0 = {}, c1 = {}, c2 = {}, c3 = {};
    for (int k0 = 0; k0 < DD; k0 += 32) {
        v16h a = load_a_frag(Arow, DD, k0, lane);
        v16h b0 = load_b_fragT(Wt, DD, tn * 64 + 0 * 16 + l15, k0, lane);
        v16h b1 = load_b_fragT(Wt, DD, tn * 64 + 1 * 16 + l15, k0, lane);
        v16h b2 = load_b_fragT(Wt, DD, tn * 64 + 2 * 16 + l15, k0, lane);
        v16h b3 = load_b_fragT(Wt, DD, tn * 64 + 3 * 16 + l15, k0, lane);
        c0 = __builtin_amdgcn_wmma_f32_16x16x32_f16(false, a, false, b0, (short)0, c0, false, false);
        c1 = __builtin_amdgcn_wmma_f32_16x16x32_f16(false, a, false, b1, (short)0, c1, false, false);
        c2 = __builtin_amdgcn_wmma_f32_16x16x32_f16(false, a, false, b2, (short)0, c2, false, false);
        c3 = __builtin_amdgcn_wmma_f32_16x16x32_f16(false, a, false, b3, (short)0, c3, false, false);
    }
    v8f cs[4] = {c0, c1, c2, c3};
#pragma unroll
    for (int j = 0; j < 4; ++j) {
        int col = tn * 64 + j * 16 + l15;
#pragma unroll
        for (int r = 0; r < 8; ++r) {
            int row = tm * 16 + r + ((lane < 16) ? 0 : 8);
            outF[(size_t)row * DD + col] = cs[j][r];
        }
    }
}

// ---------------------------------------------------------- fused attention
// grid (T/QT, B*H), 256 threads.
// LDS: wh (QT*TT f16, 32KB) | sc (QT*TT f32, 64KB) | red 512 f | fac QT f
__global__ void attn_kernel(const half_t* __restrict__ inner,   // (B,H,T,dh) f16
                            const half_t* __restrict__ outer,   // (B,H,T,dh) f16
                            const half_t* __restrict__ vmatT,   // (B,H,dh,T) f16
                            const float* __restrict__ aperture,  // (B,T,H)
                            const float* __restrict__ expression,
                            float* __restrict__ converged) {     // (B,H,T,dh) f32
    extern __shared__ char smraw[];
    half_t* wh = (half_t*)smraw;                      // QT*TT halves
    float*  sc = (float*)(smraw + QT * TT * 2);       // QT*TT floats
    float* red = sc + QT * TT;                        // 512
    float* fac = red + 512;                           // QT

    int wave = threadIdx.x >> 5, lane = threadIdx.x & 31;
    int l15  = lane & 15;
    int ko   = (lane < 16) ? 0 : 8;
    int kb0  = (lane < 16) ? 0 : 16;
    int bh   = blockIdx.y;
    int b = bh / HH, h = bh % HH;
    int q0 = blockIdx.x * QT;

    const half_t* innP = inner + ((size_t)bh * TT + q0) * DH;
    const half_t* outP = outer + (size_t)bh * TT * DH;
    const half_t* vtP  = vmatT + (size_t)bh * DH * TT;

    // ---- phase 1: scores (16 x 1024) via WMMA, strip-mined over key tiles
    for (int kt = wave; kt < TT / 16; kt += 8) {
        v8f c = {};
#pragma unroll
        for (int k0 = 0; k0 < DH; k0 += 32) {
            v16h a = load_a_frag(innP, DH, k0, lane);
            // outer rows contiguous in d -> one 32B vector load per lane
            v16h bf = *(const v16h*)(outP + (size_t)(kt * 16 + l15) * DH + k0 + kb0);
            c = __builtin_amdgcn_wmma_f32_16x16x32_f16(false, a, false, bf, (short)0, c, false, false);
        }
#pragma unroll
        for (int r = 0; r < 8; ++r) {
            int row = r + ((lane < 16) ? 0 : 8);
            sc[row * TT + kt * 16 + l15] = c[r] * 0.125f;   // 1/sqrt(dh)
        }
    }
    __syncthreads();

    // ---- phase 2: softmax + gating stats (16 threads per query row)
    int row = threadIdx.x >> 4;        // 0..15
    int seg = threadIdx.x & 15;        // 0..15, 64 cols each
    float* srow = sc + row * TT;
    float mx = -1e30f;
    for (int k = seg * 64; k < seg * 64 + 64; ++k) mx = fmaxf(mx, srow[k]);
    red[threadIdx.x] = mx;
    __syncthreads();
    float rmax = -1e30f;
#pragma unroll
    for (int j = 0; j < 16; ++j) rmax = fmaxf(rmax, red[row * 16 + j]);
    __syncthreads();                   // done reading red before reuse
    float psum = 0.0f, wsum = 0.0f;
    for (int k = seg * 64; k < seg * 64 + 64; ++k) {
        float p = __expf(srow[k] - rmax);
        float e = expression[((size_t)b * TT + k) * HH + h];
        float w = p * e;
        wh[row * TT + k] = (half_t)w;  // gated weights as f16 for WMMA phase 3
        psum += p; wsum += w;
    }
    red[threadIdx.x] = psum;
    red[256 + threadIdx.x] = wsum;
    __syncthreads();
    if (seg == 0) {
        float tp = 0.0f, tw = 0.0f;
#pragma unroll
        for (int j = 0; j < 16; ++j) { tp += red[row * 16 + j]; tw += red[256 + row * 16 + j]; }
        float aq = aperture[((size_t)b * TT + (q0 + row)) * HH + h];
        fac[row] = aq / (aq * tw + 1e-8f * tp);   // renormalization collapses to a row scalar
    }
    __syncthreads();

    // ---- phase 3: converged = (W @ V) * fac, 4 waves x one 16x16 tile
    if (wave < 4) {
        int d0 = wave * 16;
        v8f c = {};
        for (int k0 = 0; k0 < TT; k0 += 32) {
            v16h a = load_a_frag(wh, TT, k0, lane);          // ds_load_b128 x2
            // V transposed: contiguous in key index -> one 32B load per lane
            v16h bf = *(const v16h*)(vtP + (size_t)(d0 + l15) * TT + k0 + kb0);
            c = __builtin_amdgcn_wmma_f32_16x16x32_f16(false, a, false, bf, (short)0, c, false, false);
        }
#pragma unroll
        for (int r = 0; r < 8; ++r) {
            int qr = r + ((lane < 16) ? 0 : 8);
            converged[((size_t)bh * TT + q0 + qr) * DH + d0 + l15] = c[r] * fac[qr];
        }
    }
}

// ------------------------------- chamber rotation -> combined (B,T,D) f16
__global__ void chamber_kernel(const float* __restrict__ conv,    // (B,H,T,dh)
                               const float* __restrict__ chi,
                               const float* __restrict__ beta_p,
                               const float* __restrict__ iv_p,
                               const float* __restrict__ ov_p,
                               half_t* __restrict__ combined) {   // (B,T,D)
    const int half = DH / 2;
    int idx = blockIdx.x * blockDim.x + threadIdx.x;              // B*H*T*32
    if (idx >= BB * HH * TT * half) return;
    int d  = idx % half;
    int t  = (idx / half) % TT;
    int bh = idx / (half * TT);
    int b = bh / HH, h = bh % HH;

    float beta = sigmoidf_(beta_p[h]);
    float iv   = sigmoidf_(iv_p[h]);
    float ov   = sigmoidf_(ov_p[h]);
    float ang  = 3.14159265358979323846f * beta;
    float ca = cosf(ang), sa = sinf(ang);
    float th = tanhf(chi[h]);

    const float* base = conv + ((size_t)bh * TT + t) * DH;
    float xr = base[d] * iv;
    float xi = base[d + half] * iv;
    float s  = ov * th;
    size_t ob = ((size_t)b * TT + t) * DD + (size_t)h * DH;
    combined[ob + d]        = (half_t)((xr * ca - xi * sa) * s);
    combined[ob + half + d] = (half_t)((xr * sa + xi * ca) * s);
}

// ---------------------------------------------------------------------------
extern "C" void kernel_launch(void* const* d_in, const int* in_sizes, int n_in,
                              void* d_out, int out_size, void* d_ws, size_t ws_size,
                              hipStream_t stream) {
    const float* x      = (const float*)d_in[0];
    const float* Wa     = (const float*)d_in[1];
    const float* ba     = (const float*)d_in[2];
    const float* We     = (const float*)d_in[3];
    const float* be     = (const float*)d_in[4];
    const float* Wi     = (const float*)d_in[5];
    const float* Wo     = (const float*)d_in[6];
    const float* Wv     = (const float*)d_in[7];
    const float* Wout   = (const float*)d_in[8];
    const float* chi    = (const float*)d_in[9];
    const float* beta_p = (const float*)d_in[10];
    const float* iv_p   = (const float*)d_in[11];
    const float* ov_p   = (const float*)d_in[12];
    float* out = (float*)d_out;

    // workspace carve-up
    char* ws = (char*)d_ws;
    half_t* xh       = (half_t*)ws;                 ws += (size_t)MM * DD * 2;   // 8 MB
    half_t* WiT_h    = (half_t*)ws;                 ws += (size_t)DD * DD * 2;   // transposed
    half_t* WoT_h    = (half_t*)ws;                 ws += (size_t)DD * DD * 2;
    half_t* WvT_h    = (half_t*)ws;                 ws += (size_t)DD * DD * 2;
    half_t* WoutT_h  = (half_t*)ws;                 ws += (size_t)DD * DD * 2;
    half_t* inner_h  = (half_t*)ws;                 ws += (size_t)MM * DD * 2;   // (B,H,T,dh)
    half_t* outer_h  = (half_t*)ws;                 ws += (size_t)MM * DD * 2;   // (B,H,T,dh)
    half_t* vT_h     = (half_t*)ws;                 ws += (size_t)MM * DD * 2;   // (B,H,dh,T)
    float*  aper     = (float*)ws;                  ws += (size_t)MM * HH * 4;
    float*  expr     = (float*)ws;                  ws += (size_t)MM * HH * 4;
    float*  convg    = (float*)ws;                  ws += (size_t)MM * DD * 4;   // 16 MB
    half_t* comb_h   = (half_t*)ws;                 ws += (size_t)MM * DD * 2;

    const int nX = MM * DD, nW = DD * DD;
    f2h_kernel  <<<(nX + 255) / 256, 256, 0, stream>>>(x, xh, nX);
    f2h_T_kernel<<<(nW + 255) / 256, 256, 0, stream>>>(Wi,   WiT_h,   DD, DD);
    f2h_T_kernel<<<(nW + 255) / 256, 256, 0, stream>>>(Wo,   WoT_h,   DD, DD);
    f2h_T_kernel<<<(nW + 255) / 256, 256, 0, stream>>>(Wv,   WvT_h,   DD, DD);
    f2h_T_kernel<<<(nW + 255) / 256, 256, 0, stream>>>(Wout, WoutT_h, DD, DD);

    gates_kernel<<<(MM * HH + 255) / 256, 256, 0, stream>>>(x, Wa, ba, We, be, aper, expr);

    const int tiles = (MM / 16) * (DD / 64);        // 4096 strips of 16x64
    const int gemmBlocks = tiles / 8;               // 512 blocks of 8 waves
    gemm16x64_heads_kernel<<<gemmBlocks, 256, 0, stream>>>(xh, WiT_h, inner_h, 0);
    gemm16x64_heads_kernel<<<gemmBlocks, 256, 0, stream>>>(xh, WoT_h, outer_h, 0);
    gemm16x64_heads_kernel<<<gemmBlocks, 256, 0, stream>>>(xh, WvT_h, vT_h,    1);

    size_t smemBytes = (size_t)QT * TT * 2 + (size_t)(QT * TT + 512 + QT) * sizeof(float); // ~98 KB
    dim3 agrid(TT / QT, BB * HH);
    attn_kernel<<<agrid, 256, smemBytes, stream>>>(inner_h, outer_h, vT_h, aper, expr, convg);

    const int nPairs = BB * HH * TT * (DH / 2);
    chamber_kernel<<<(nPairs + 255) / 256, 256, 0, stream>>>(convg, chi, beta_p, iv_p, ov_p, comb_h);

    gemm16x64_out_kernel<<<gemmBlocks, 256, 0, stream>>>(comb_h, WoutT_h, out);
}
